// EHEMB_66735201845492
// MI455X (gfx1250) — matrix operand order repaired and checked by reading, not compiled, not run on hardware
//
#include <hip/hip_runtime.h>

// ---------------------------------------------------------------------------
// Model dims
// ---------------------------------------------------------------------------
#define D_NINP 552
#define D_NHID 300
#define D_NTOK 256
#define D_L    4096
#define D_HD   138
#define D_HDP  160      // HD padded to 5*32 for WMMA K-steps
#define D_NH   4
#define D_KC   1104     // 2*NINP (conv-as-GEMM K)
// padded dims (multiples of 32 for K, 64 for N)
#define P_552  576
#define P_300  320
#define P_1104 1120

typedef __attribute__((ext_vector_type(16))) __bf16 v16bf;
typedef __attribute__((ext_vector_type(8)))  float  v8f;
typedef unsigned short ushort_t;

// POD 16-byte chunk (HIP's uint4 has user ctors -> can't live in a union)
struct __attribute__((aligned(16))) U4 { unsigned a[4]; };

union Frag {
    v16bf          v;
    unsigned       u[8];
    U4             q[2];
    unsigned short h16[16];
};

union Pack8 {
    U4             u4;
    unsigned short h[8];
};

// float -> bf16 (round to nearest even), raw bits
__device__ __forceinline__ unsigned short f2bf(float f) {
    unsigned u = __float_as_uint(f);
    unsigned r = u + 0x7FFFu + ((u >> 16) & 1u);
    return (unsigned short)(r >> 16);
}

// A-matrix 16x32 bf16 element->K map: VGPR r (pair base), lane half h
__device__ __forceinline__ int kkA(int r, int h) {
    return (r < 4) ? (8 * h + 2 * r) : (16 + 8 * h + 2 * (r - 4));
}

// ---------------------------------------------------------------------------
// WMMA GEMM on pre-padded bf16 operands.
//   A: [M][lda] bf16, zero-padded K region (lda >= Kp, lda*2 % 16 == 0)
//   B: [Np][ldb] bf16, TRANSPOSED (n-major, K contiguous), zero-padded
//   C = A @ B^T + bias
// block = 256 threads (8 waves), tile 128(M) x 64(N), K-step 32.
// mode: 0 f32 out (stride N) ; 1 f32+ReLU ; 2 bf16 head-padded [h][row][160]
//       3 bf16 out (stride ldc) ; 4 bf16+ReLU (stride ldc)
//       5 bf16 transposed head layout [h][d][4096] (for V), packed b128 stores
// ---------------------------------------------------------------------------
__global__ __launch_bounds__(256) void gemm_wmma_kernel(
        const ushort_t* __restrict__ A, int lda,
        const ushort_t* __restrict__ B, int ldb,
        const float* __restrict__ bias,
        void* __restrict__ out,
        int N, int Kp, int mode, int ldc)
{
    __shared__ __attribute__((aligned(16))) unsigned short lA[128 * 40];
    __shared__ __attribute__((aligned(16))) unsigned short lB[64 * 40];

    const int tid  = threadIdx.x;
    const int w    = tid >> 5;
    const int lane = tid & 31;
    const int half = lane >> 4;
    const int l15  = lane & 15;
    const int mBase = blockIdx.y * 128;
    const int nBase = blockIdx.x * 64;

    const v8f vz = {};
    v8f acc[4];
    acc[0] = vz; acc[1] = vz; acc[2] = vz; acc[3] = vz;

    // staging maps (all b128, branch-free)
    const int arow  = tid >> 1;        // 0..127
    const int akoff = (tid & 1) * 16;  // 0 / 16
    const int brow  = tid >> 2;        // 0..63
    const int bkoff = (tid & 3) * 8;   // 0..24

    const ushort_t* apg = A + (size_t)(mBase + arow) * lda + akoff;
    const ushort_t* bpg = B + (size_t)(nBase + brow) * ldb + bkoff;

    for (int kt = 0; kt < Kp; kt += 32) {
        __builtin_prefetch(apg + kt + 32, 0, 3);   // global_prefetch_b8
        {
            const U4* s = (const U4*)(apg + kt);
            U4* d = (U4*)(&lA[arow * 40 + akoff]);
            d[0] = s[0];
            d[1] = s[1];
        }
        {
            const U4* s = (const U4*)(bpg + kt);
            U4* d = (U4*)(&lB[brow * 40 + bkoff]);
            d[0] = s[0];
        }
        __syncthreads();

        Frag af;
        const int ar40 = (w * 16 + l15) * 40;
        #pragma unroll
        for (int r = 0; r < 8; r++)
            af.u[r] = *(const unsigned*)(&lA[ar40 + 16 * half + 2 * r]);

        #pragma unroll
        for (int nt = 0; nt < 4; nt++) {
            Frag bf;
            const int c40 = (nt * 16 + l15) * 40 + 16 * half;
            #pragma unroll
            for (int r = 0; r < 8; r++)
                bf.u[r] = *(const unsigned*)(&lB[c40 + 2 * r]);
            acc[nt] = __builtin_amdgcn_wmma_f32_16x16x32_bf16(
                false, af.v, false, bf.v, (short)0, acc[nt], false, false);
        }
        __syncthreads();
    }

    // ---- epilogue ----
    const int row0 = mBase + w * 16 + 8 * half;   // first of 8 consecutive rows
    #pragma unroll
    for (int nt = 0; nt < 4; nt++) {
        const int col = nBase + nt * 16 + l15;
        if (col >= N) continue;
        const float bb = bias[col];
        if (mode == 5) {
            // V transposed: [head][d][4096], 8 consecutive rows -> one b128 store
            const int hh = col / D_HD, d = col - hh * D_HD;
            Pack8 pk;
            #pragma unroll
            for (int r = 0; r < 8; r++) pk.h[r] = f2bf(acc[nt][r] + bb);
            *(U4*)((ushort_t*)out + ((size_t)(hh * D_HDP + d)) * D_L + row0) = pk.u4;
        } else {
            #pragma unroll
            for (int r = 0; r < 8; r++) {
                const int row = row0 + r;
                float v = acc[nt][r] + bb;
                if (mode == 1 || mode == 4) v = fmaxf(v, 0.0f);
                if (mode == 2) {
                    int hh = col / D_HD, d = col - hh * D_HD;
                    ((ushort_t*)out)[((size_t)hh * D_L + row) * D_HDP + d] = f2bf(v);
                } else if (mode >= 3) {
                    ((ushort_t*)out)[(size_t)row * ldc + col] = f2bf(v);
                } else {
                    ((float*)out)[(size_t)row * N + col] = v;
                }
            }
        }
    }
}

// ---------------------------------------------------------------------------
// Flash attention: 1 wave = 16 queries of one head; 32-key blocks.
// Q/K bf16 head-padded [4][4096][160]; V bf16 TRANSPOSED [4][160][4096].
// Output bf16 [4096][576] (pre-zeroed pad cols).
// ---------------------------------------------------------------------------
__global__ __launch_bounds__(128) void attn_kernel(
        const ushort_t* __restrict__ Qh,
        const ushort_t* __restrict__ Kh,
        const ushort_t* __restrict__ Vt,
        ushort_t* __restrict__ AOb)
{
    __shared__ __attribute__((aligned(16))) unsigned short lP[4][16 * 32];

    const int w    = threadIdx.x >> 5;
    const int lane = threadIdx.x & 31;
    const int half = lane >> 4;
    const int l15  = lane & 15;
    const int head  = blockIdx.y;
    const int qbase = (blockIdx.x * 4 + w) * 16;
    const size_t hbase = (size_t)head * D_L * D_HDP;

    unsigned short* lPw = lP[w];

    Frag qf[5];
    const ushort_t* qp = Qh + hbase + (size_t)(qbase + l15) * D_HDP;
    #pragma unroll
    for (int ds = 0; ds < 5; ds++) {
        qf[ds].q[0] = *(const U4*)(qp + ds * 32 + 16 * half);
        qf[ds].q[1] = *(const U4*)(qp + ds * 32 + 16 * half + 8);
    }

    float mrow[8], lrow[8];
    #pragma unroll
    for (int r = 0; r < 8; r++) { mrow[r] = -1e30f; lrow[r] = 0.0f; }
    const v8f vz = {};
    v8f ofr[10];
    #pragma unroll
    for (int dt = 0; dt < 10; dt++) ofr[dt] = vz;

    const float scale = 0.08512565307587486f;  // 1/sqrt(138)
    // per-lane V base: column d = dt*16 + l15, keys contiguous
    const ushort_t* vbase = Vt + hbase + (size_t)l15 * D_L + 16 * half;

    for (int kb = 0; kb < D_L; kb += 32) {
        // S = Q @ K^T  (two 16x16 key tiles)
        v8f s0 = vz, s1 = vz;
        const ushort_t* kp0 = Kh + hbase + (size_t)(kb + l15) * D_HDP + 16 * half;
        const ushort_t* kp1 = kp0 + 16 * D_HDP;
        #pragma unroll
        for (int ds = 0; ds < 5; ds++) {
            Frag b0, b1;
            b0.q[0] = *(const U4*)(kp0 + ds * 32);
            b0.q[1] = *(const U4*)(kp0 + ds * 32 + 8);
            b1.q[0] = *(const U4*)(kp1 + ds * 32);
            b1.q[1] = *(const U4*)(kp1 + ds * 32 + 8);
            s0 = __builtin_amdgcn_wmma_f32_16x16x32_bf16(
                false, qf[ds].v, false, b0.v, (short)0, s0, false, false);
            s1 = __builtin_amdgcn_wmma_f32_16x16x32_bf16(
                false, qf[ds].v, false, b1.v, (short)0, s1, false, false);
        }

        // online softmax (rows in lane halves; reduce over 16 lanes)
        float p0[8], p1[8], corr[8];
        #pragma unroll
        for (int r = 0; r < 8; r++) {
            float a0 = s0[r] * scale, a1 = s1[r] * scale;
            float mx = fmaxf(a0, a1);
            mx = fmaxf(mx, __shfl_xor(mx, 1, 32));
            mx = fmaxf(mx, __shfl_xor(mx, 2, 32));
            mx = fmaxf(mx, __shfl_xor(mx, 4, 32));
            mx = fmaxf(mx, __shfl_xor(mx, 8, 32));
            float mn = fmaxf(mrow[r], mx);
            float c  = __expf(mrow[r] - mn);
            mrow[r] = mn; corr[r] = c;
            p0[r] = __expf(a0 - mn);
            p1[r] = __expf(a1 - mn);
            float rs = p0[r] + p1[r];
            rs += __shfl_xor(rs, 1, 32);
            rs += __shfl_xor(rs, 2, 32);
            rs += __shfl_xor(rs, 4, 32);
            rs += __shfl_xor(rs, 8, 32);
            lrow[r] = lrow[r] * c + rs;
        }
        #pragma unroll
        for (int dt = 0; dt < 10; dt++) {
            #pragma unroll
            for (int r = 0; r < 8; r++) ofr[dt][r] *= corr[r];
        }

        // C-layout -> A-layout for P via per-wave LDS tile
        #pragma unroll
        for (int r = 0; r < 8; r++) {
            lPw[(8 * half + r) * 32 + l15]      = f2bf(p0[r]);
            lPw[(8 * half + r) * 32 + 16 + l15] = f2bf(p1[r]);
        }
        __builtin_amdgcn_wave_barrier();
        Frag pf;
        #pragma unroll
        for (int r = 0; r < 8; r++)
            pf.u[r] = *(const unsigned*)(&lPw[l15 * 32 + 16 * half + 2 * r]);
        __builtin_amdgcn_wave_barrier();

        // O += P @ V : B frags are contiguous key runs in transposed V
        #pragma unroll
        for (int dt = 0; dt < 10; dt++) {
            Frag bv;
            const ushort_t* vp = vbase + (size_t)dt * 16 * D_L + kb;
            bv.q[0] = *(const U4*)(vp);
            bv.q[1] = *(const U4*)(vp + 8);
            ofr[dt] = __builtin_amdgcn_wmma_f32_16x16x32_bf16(
                false, pf.v, false, bv.v, (short)0, ofr[dt], false, false);
        }
    }

    // finalize -> bf16 [4096][576]
    float inv[8];
    #pragma unroll
    for (int r = 0; r < 8; r++) inv[r] = 1.0f / lrow[r];
    #pragma unroll
    for (int dt = 0; dt < 10; dt++) {
        const int d = dt * 16 + l15;
        if (d >= D_HD) continue;
        #pragma unroll
        for (int r = 0; r < 8; r++) {
            const int row = qbase + 8 * half + r;
            AOb[(size_t)row * P_552 + head * D_HD + d] = f2bf(ofr[dt][r] * inv[r]);
        }
    }
}

// ---------------------------------------------------------------------------
// LayerNorm: LN(a + b); f32 out optional, bf16 out (self-padding) mandatory
// ---------------------------------------------------------------------------
__global__ __launch_bounds__(256) void ln_kernel(
        const float* __restrict__ a, int lda,
        const float* __restrict__ b, int ldb,
        const float* __restrict__ sc, const float* __restrict__ bi,
        float* __restrict__ outf,
        ushort_t* __restrict__ outb, int ldout)
{
    const int w = threadIdx.x >> 5, lane = threadIdx.x & 31;
    const int row = blockIdx.x * 8 + w;
    const float* ar = a + (size_t)row * lda;
    const float* br = b + (size_t)row * ldb;
    float va[18];
    float s = 0.0f, s2 = 0.0f;
    int idx = 0;
    for (int c = lane; c < D_NINP; c += 32) {
        float v = ar[c] + br[c];
        va[idx++] = v; s += v; s2 += v * v;
    }
    for (int m = 1; m < 32; m <<= 1) {
        s  += __shfl_xor(s,  m, 32);
        s2 += __shfl_xor(s2, m, 32);
    }
    const float mean = s * (1.0f / D_NINP);
    const float var  = s2 * (1.0f / D_NINP) - mean * mean;
    const float rs   = rsqrtf(var + 1e-5f);
    ushort_t* ob = outb + (size_t)row * ldout;
    float* of = outf ? outf + (size_t)row * D_NINP : (float*)0;
    idx = 0;
    for (int c = lane; c < D_NINP; c += 32) {
        float v = (va[idx++] - mean) * rs * sc[c] + bi[c];
        ob[c] = f2bf(v);
        if (of) of[c] = v;
    }
    for (int c = D_NINP + lane; c < ldout; c += 32) ob[c] = 0;
}

// ---------------------------------------------------------------------------
// Prep kernels (fp32 inputs -> padded bf16 operands)
// ---------------------------------------------------------------------------
__global__ void prep_xc_kernel(const float* __restrict__ Fia,
                               const int* __restrict__ xi1,
                               const float* __restrict__ emb,
                               ushort_t* __restrict__ Xc)
{
    const long total = (long)D_L * P_1104;
    for (long i = (long)blockIdx.x * blockDim.x + threadIdx.x; i < total;
         i += (long)gridDim.x * blockDim.x) {
        int l = (int)(i / P_1104), j = (int)(i % P_1104);
        unsigned short o = 0;
        if (j < D_KC) {
            int k = j / D_NINP, c = j - k * D_NINP;
            int t = 2 * l + k;
            float v = (t < D_L) ? emb[(size_t)xi1[t] * D_NINP + c]
                                : Fia[((size_t)2 * t - 2 * D_L) * D_NINP + c];
            o = f2bf(v);
        }
        Xc[i] = o;
    }
}

__global__ void prep_fia2_kernel(const float* __restrict__ Fia, ushort_t* __restrict__ O)
{
    const long total = (long)D_L * P_552;
    for (long i = (long)blockIdx.x * blockDim.x + threadIdx.x; i < total;
         i += (long)gridDim.x * blockDim.x) {
        int l = (int)(i / P_552), c = (int)(i % P_552);
        O[i] = (c < D_NINP) ? f2bf(Fia[((size_t)2 * l + 1) * D_NINP + c]) : (unsigned short)0;
    }
}

// dense weight [K][N] f32 -> transposed padded bf16 [Np][Kp]
__global__ void prep_wt_kernel(const float* __restrict__ wsrc, ushort_t* __restrict__ O,
                               int K, int N, int Kp, int Np)
{
    const long total = (long)Np * Kp;
    for (long i = (long)blockIdx.x * blockDim.x + threadIdx.x; i < total;
         i += (long)gridDim.x * blockDim.x) {
        int n = (int)(i / Kp), k = (int)(i % Kp);
        O[i] = (n < N && k < K) ? f2bf(wsrc[(size_t)k * N + n]) : (unsigned short)0;
    }
}

// conv weight (o,c,k) -> transposed padded bf16 [576][1120], kk = k*552+c
__global__ void prep_wconv_kernel(const float* __restrict__ wsrc, ushort_t* __restrict__ O)
{
    const long total = (long)P_552 * P_1104;
    for (long i = (long)blockIdx.x * blockDim.x + threadIdx.x; i < total;
         i += (long)gridDim.x * blockDim.x) {
        int n = (int)(i / P_1104), kk = (int)(i % P_1104);
        unsigned short o = 0;
        if (n < D_NINP && kk < D_KC) {
            int k = kk / D_NINP, c = kk - k * D_NINP;
            o = f2bf(wsrc[(size_t)n * D_KC + c * 2 + k]);
        }
        O[i] = o;
    }
}

// A3[l][o] = Fia2[l][o] + (l<2048 ? P2top[l][o] : p2_b[o]) -> bf16 [4096][576]
__global__ void build_a3_kernel(const float* __restrict__ Fia,
                                const float* __restrict__ P2,
                                const float* __restrict__ p2b,
                                ushort_t* __restrict__ A3)
{
    const long total = (long)D_L * P_552;
    for (long i = (long)blockIdx.x * blockDim.x + threadIdx.x; i < total;
         i += (long)gridDim.x * blockDim.x) {
        int l = (int)(i / P_552), o = (int)(i % P_552);
        unsigned short out = 0;
        if (o < D_NINP) {
            float v = Fia[((size_t)2 * l + 1) * D_NINP + o];
            v += (l < D_L / 2) ? P2[(size_t)l * D_NINP + o] : p2b[o];
            out = f2bf(v);
        }
        A3[i] = out;
    }
}

__global__ void zero_kernel(U4* __restrict__ p, long n)
{
    U4 z; z.a[0] = z.a[1] = z.a[2] = z.a[3] = 0u;
    for (long i = (long)blockIdx.x * blockDim.x + threadIdx.x; i < n;
         i += (long)gridDim.x * blockDim.x)
        p[i] = z;
}

// ---------------------------------------------------------------------------
// Host orchestration
// ---------------------------------------------------------------------------
extern "C" void kernel_launch(void* const* d_in, const int* in_sizes, int n_in,
                              void* d_out, int out_size, void* d_ws, size_t ws_size,
                              hipStream_t stream)
{
    (void)in_sizes; (void)n_in; (void)out_size; (void)ws_size;

    const float* Fia   = (const float*)d_in[0];
    const int*   xi1   = (const int*)  d_in[1];
    const float* emb   = (const float*)d_in[2];
    const float* c1_w  = (const float*)d_in[3];
    const float* c1_b  = (const float*)d_in[4];
    const float* wq    = (const float*)d_in[5];
    const float* bq    = (const float*)d_in[6];
    const float* wk    = (const float*)d_in[7];
    const float* bk    = (const float*)d_in[8];
    const float* wv    = (const float*)d_in[9];
    const float* bv    = (const float*)d_in[10];
    const float* wo    = (const float*)d_in[11];
    const float* bo    = (const float*)d_in[12];
    const float* ln1_s = (const float*)d_in[13];
    const float* ln1_b = (const float*)d_in[14];
    const float* f1_w  = (const float*)d_in[15];
    const float* f1_b  = (const float*)d_in[16];
    const float* f2_w  = (const float*)d_in[17];
    const float* f2_b  = (const float*)d_in[18];
    const float* ln2_s = (const float*)d_in[19];
    const float* ln2_b = (const float*)d_in[20];
    const float* p2_w  = (const float*)d_in[21];
    const float* p2_b  = (const float*)d_in[22];
    const float* dec_w = (const float*)d_in[23];
    const float* dec_b = (const float*)d_in[24];

    // ---- workspace layout (bytes), liveness-reused, total ~51.2 MB ----
    char* W = (char*)d_ws;
    ushort_t* Xc    = (ushort_t*)(W + 0);          // [4096][1120] bf16   9,175,040
    ushort_t* Fia2b = (ushort_t*)(W + 9175040);    // [4096][576]  bf16   4,718,592
    ushort_t* Fi1b  = (ushort_t*)(W + 13893632);   // [4096][576]  bf16 (later Hb)
    ushort_t* Qh    = (ushort_t*)(W + 18612224);   // 3x [4][4096][160]  15,728,640
    ushort_t* Kh    = Qh + (size_t)D_NH * D_L * D_HDP;
    ushort_t* Vt    = Kh + (size_t)D_NH * D_L * D_HDP;  // transposed [4][160][4096]
    float*    Hf    = (float*)(W + 34340864);      // [4096][552] f32 (later P2t)
    float*    T1    = (float*)(W + 43384832);      // [4096][552] f32 (T1/T2)
    ushort_t* Wb    = (ushort_t*)(W + 52428800);   // weight scratch 1,290,240
    // reuse:
    ushort_t* AOb  = Xc;                            // [4096][576] bf16
    ushort_t* FF1b = (ushort_t*)(W + 4718592);      // [4096][320] bf16 (in Xc region)
    ushort_t* Hb   = Fi1b;                          // [4096][576] bf16
    ushort_t* A3b  = Qh;                            // [4096][576] bf16
    ushort_t* H2b  = (ushort_t*)(W + 18612224 + 4718592); // [4096][552]+guard
    float*    P2t  = Hf;                            // [2048][552] f32
    float*    T2   = T1;

    // ---- 1) conv1 as GEMM: Fi1 = Xc @ Wc1^T + c1_b ----
    prep_xc_kernel<<<2048, 256, 0, stream>>>(Fia, xi1, emb, Xc);
    prep_wconv_kernel<<<1024, 256, 0, stream>>>(c1_w, Wb);
    zero_kernel<<<1024, 256, 0, stream>>>((U4*)Fi1b, (long)(D_L * P_552 / 8));
    gemm_wmma_kernel<<<dim3(9, 32), 256, 0, stream>>>(
        Xc, P_1104, Wb, P_1104, c1_b, Fi1b, D_NINP, P_1104, 3, P_552);

    // ---- 2) QKV (Q/K head-padded, V transposed) ----
    prep_fia2_kernel<<<2048, 256, 0, stream>>>(Fia, Fia2b);
    zero_kernel<<<1024, 256, 0, stream>>>((U4*)Qh, (long)(3 * D_NH * D_L * D_HDP / 8));
    prep_wt_kernel<<<1024, 256, 0, stream>>>(wq, Wb, D_NINP, D_NINP, P_552, P_552);
    gemm_wmma_kernel<<<dim3(9, 32), 256, 0, stream>>>(
        Fia2b, P_552, Wb, P_552, bq, Qh, D_NINP, P_552, 2, 0);
    prep_wt_kernel<<<1024, 256, 0, stream>>>(wk, Wb, D_NINP, D_NINP, P_552, P_552);
    gemm_wmma_kernel<<<dim3(9, 32), 256, 0, stream>>>(
        Fi1b, P_552, Wb, P_552, bk, Kh, D_NINP, P_552, 2, 0);
    prep_wt_kernel<<<1024, 256, 0, stream>>>(wv, Wb, D_NINP, D_NINP, P_552, P_552);
    gemm_wmma_kernel<<<dim3(9, 32), 256, 0, stream>>>(
        Fi1b, P_552, Wb, P_552, bv, Vt, D_NINP, P_552, 5, 0);

    // ---- 3) flash attention -> AOb bf16 [4096][576] ----
    zero_kernel<<<1024, 256, 0, stream>>>((U4*)AOb, (long)(D_L * P_552 / 8));
    attn_kernel<<<dim3(64, D_NH), 128, 0, stream>>>(Qh, Kh, Vt, AOb);

    // ---- 4) output proj + residual + LN1 ----
    prep_wt_kernel<<<1024, 256, 0, stream>>>(wo, Wb, D_NINP, D_NINP, P_552, P_552);
    gemm_wmma_kernel<<<dim3(9, 32), 256, 0, stream>>>(
        AOb, P_552, Wb, P_552, bo, T1, D_NINP, P_552, 0, 0);
    ln_kernel<<<512, 256, 0, stream>>>(T1, D_NINP, Fia + D_NINP, 2 * D_NINP,
                                       ln1_s, ln1_b, Hf, Hb, P_552);

    // ---- 5) FFN + residual + LN2 ----
    prep_wt_kernel<<<1024, 256, 0, stream>>>(f1_w, Wb, D_NINP, D_NHID, P_552, P_300);
    zero_kernel<<<1024, 256, 0, stream>>>((U4*)FF1b, (long)(D_L * P_300 / 8));
    gemm_wmma_kernel<<<dim3(5, 32), 256, 0, stream>>>(
        Hb, P_552, Wb, P_552, f1_b, FF1b, D_NHID, P_552, 4, P_300);   // ReLU
    prep_wt_kernel<<<1024, 256, 0, stream>>>(f2_w, Wb, D_NHID, D_NINP, P_300, P_552);
    gemm_wmma_kernel<<<dim3(9, 32), 256, 0, stream>>>(
        FF1b, P_300, Wb, P_300, f2_b, T2, D_NINP, P_300, 0, 0);
    zero_kernel<<<1, 32, 0, stream>>>((U4*)(H2b + (size_t)D_L * D_NINP), 4); // guard zone
    ln_kernel<<<512, 256, 0, stream>>>(T2, D_NINP, Hf, D_NINP,
                                       ln2_s, ln2_b, (float*)0, H2b, D_NINP);

    // ---- 6) p2 conv: pairs of H2 rows are contiguous -> lda=1104 ----
    prep_wconv_kernel<<<1024, 256, 0, stream>>>(p2_w, Wb);
    gemm_wmma_kernel<<<dim3(9, 16), 256, 0, stream>>>(
        H2b, D_KC, Wb, P_1104, p2_b, P2t, D_NINP, P_1104, 0, 0);
    build_a3_kernel<<<2048, 256, 0, stream>>>(Fia, P2t, p2_b, A3b);

    // ---- 7) decoder -> d_out (4096 x 256 f32) ----
    prep_wt_kernel<<<1024, 256, 0, stream>>>(dec_w, Wb, D_NINP, D_NTOK, P_552, D_NTOK);
    gemm_wmma_kernel<<<dim3(4, 32), 256, 0, stream>>>(
        A3b, P_552, Wb, P_552, dec_b, (float*)d_out, D_NTOK, P_552, 0, 0);
}
